// SpanMarkConv_5995774345605
// MI455X (gfx1250) — matrix operand order repaired and checked by reading, not compiled, not run on hardware
//
#include <hip/hip_runtime.h>
#include <hip/hip_bf16.h>
#include <stdint.h>

#define B_   4
#define L_   512
#define D_   768
#define W_   12
#define BL   (B_ * L_)      // 2048
#define ROWS (BL * W_)      // 24576
#define D2   (2 * D_)       // 1536
#define K3   (3 * D_)       // 2304

typedef __bf16 bf16t;
typedef __attribute__((ext_vector_type(16))) bf16t v16bf;
typedef __attribute__((ext_vector_type(8)))  float v8f;

struct __align__(16) U4 { unsigned int x, y, z, w; };
union Frag { v16bf v; U4 q[2]; };

__device__ __forceinline__ unsigned short f2bf(float f) {
  unsigned int u = __float_as_uint(f);
  u += 0x7FFFu + ((u >> 16) & 1u);   // round-to-nearest-even
  return (unsigned short)(u >> 16);
}

// A fragment (16x32 bf16, M x K): lanes 0-15 = rows; hi selects K sub-chunks
__device__ __forceinline__ void loadA(Frag& f, const unsigned short* rowp, int kbase, int hi) {
  const unsigned short* p = rowp + kbase + hi * 8;
  f.q[0] = *(const U4*)p;
  f.q[1] = *(const U4*)(p + 16);
}
// B fragment (32x16 bf16 stored N x K row-major): lane lo = column N
__device__ __forceinline__ void loadBs(Frag& f, const unsigned short* rowp, int kbase, int hi) {
  const unsigned short* p = rowp + kbase + hi * 8;
  f.q[0] = *(const U4*)p;
  f.q[1] = *(const U4*)(p + 16);
}

#define WMMA(acc, a, b) \
  acc = __builtin_amdgcn_wmma_f32_16x16x32_bf16(false, (a).v, false, (b).v, (short)0, acc, false, false)

// -------- CDNA5 async global->LDS path (guarded; falls back to register copy)
#if defined(__has_builtin)
#if __has_builtin(__builtin_amdgcn_global_load_async_to_lds_b128) && \
    __has_builtin(__builtin_amdgcn_s_wait_asynccnt)
#define USE_ASYNC 1
#endif
#endif

#ifdef USE_ASYNC
typedef int gv4i __attribute__((vector_size(16)));   // matches builtin pointee type
__device__ __forceinline__ void cpB(unsigned short* dst, const unsigned short* src) {
  __builtin_amdgcn_global_load_async_to_lds_b128(
      (__attribute__((address_space(1))) gv4i*)(src),
      (__attribute__((address_space(3))) gv4i*)(dst), 0, 0);
}
__device__ __forceinline__ void asyncWait1() { __builtin_amdgcn_s_wait_asynccnt(1); }
__device__ __forceinline__ void asyncWait0() { __builtin_amdgcn_s_wait_asynccnt(0); }
// Raw split workgroup barrier: no implicit vmem drain (LDS ordering is handled
// explicitly by s_wait_asynccnt; A loads are waited at their consuming WMMA).
__device__ __forceinline__ void wgBarrier() {
  asm volatile("s_barrier_signal -1\n\ts_barrier_wait -1" ::: "memory");
}
#else
__device__ __forceinline__ void cpB(unsigned short* dst, const unsigned short* src) {
  *(U4*)dst = *(const U4*)src;         // global_load + ds_store fallback
}
__device__ __forceinline__ void asyncWait1() {}
__device__ __forceinline__ void asyncWait0() {}
__device__ __forceinline__ void wgBarrier() { __syncthreads(); }  // needs LDS fence
#endif

// ---------------------------------------------------------------- converts
__global__ void cvt_f32_bf16(const float* __restrict__ src, unsigned short* __restrict__ dst, int n) {
  for (int i = blockIdx.x * blockDim.x + threadIdx.x; i < n; i += gridDim.x * blockDim.x)
    dst[i] = f2bf(src[i]);
}

// conv_w (o,c,k) fp32 -> (k,o,c) bf16 so each k-slice is an N x K B-matrix
__global__ void cvt_convw(const float* __restrict__ src, unsigned short* __restrict__ dst) {
  const int n = W_ * D_ * D_;
  for (int i = blockIdx.x * blockDim.x + threadIdx.x; i < n; i += gridDim.x * blockDim.x) {
    int k = i / (D_ * D_);
    int rem = i % (D_ * D_);
    dst[i] = f2bf(src[rem * W_ + k]);  // rem = o*D + c
  }
}

// -------------------------------------------------------------- GEMM frame
// Block = 256 M-rows x 64 N-cols; 8 waves x (32M x 64N).  B panel (64x32 per
// 32-K step) staged to LDS with async copies: 3 buffers, prefetch distance 2.
#define LDB_LD 40                       // padded LDS row stride (elements)
#define PANEL  (64 * LDB_LD)            // one staged B panel (elements)

struct ASet { Frag a0, a1; };

// B frags for the *current* step live in bf0..bf3 (loaded last iteration)
#define DSLOAD_B(rb) do { \
    loadBs(bf0, (rb) + (size_t)(lo) * LDB_LD,      0, hi); \
    loadBs(bf1, (rb) + (size_t)(16 + lo) * LDB_LD, 0, hi); \
    loadBs(bf2, (rb) + (size_t)(32 + lo) * LDB_LD, 0, hi); \
    loadBs(bf3, (rb) + (size_t)(48 + lo) * LDB_LD, 0, hi); } while (0)

#define WMMA8(AS_) do { \
    WMMA(acc0, (AS_).a0, bf0); WMMA(acc1, (AS_).a0, bf1); \
    WMMA(acc2, (AS_).a0, bf2); WMMA(acc3, (AS_).a0, bf3); \
    WMMA(acc4, (AS_).a1, bf0); WMMA(acc5, (AS_).a1, bf1); \
    WMMA(acc6, (AS_).a1, bf2); WMMA(acc7, (AS_).a1, bf3); } while (0)

// ------------------------------------------------- proj = relu(h @ Wp^T + b)
__global__ __launch_bounds__(256) void gemm_proj(
    const unsigned short* __restrict__ A,   // BL x D
    const unsigned short* __restrict__ Bw,  // D2 x D (N x K)
    const float* __restrict__ bias,         // D2
    unsigned short* __restrict__ out)       // BL x D2 relu'd bf16
{
  __shared__ unsigned short Bsh[3][PANEL];                 // 3 x 5 KB
  const int NB = D2 / 64;                                  // 24 N-blocks
  int nb = blockIdx.x % NB, mb = blockIdx.x / NB;
  int Nb = nb * 64;
  int w  = threadIdx.x >> 5;
  int lane = threadIdx.x & 31, lo = lane & 15, hi = lane >> 4;
  int Mb = mb * 256 + w * 32;

  const unsigned short* r0 = A + (size_t)(Mb + lo) * D_;
  const unsigned short* r1 = A + (size_t)(Mb + 16 + lo) * D_;

  int srow = threadIdx.x >> 2, schunk = threadIdx.x & 3;   // staging map
  const unsigned short* bsrc = Bw + (size_t)(Nb + srow) * D_ + schunk * 8;
  unsigned short* stBase = &Bsh[0][0] + srow * LDB_LD + schunk * 8;

  v8f acc0={},acc1={},acc2={},acc3={},acc4={},acc5={},acc6={},acc7={};
  Frag bf0, bf1, bf2, bf3;
  ASet A0;
  const int STEPS = D_ / 32;                               // 24

  cpB(stBase + 0 * PANEL, bsrc);                           // stage step 0
  cpB(stBase + 1 * PANEL, bsrc + 32);                      // stage step 1
  asyncWait1();                                            // step 0 landed
  wgBarrier();
  DSLOAD_B(&Bsh[0][0]);
  loadA(A0.a0, r0, 0, hi); loadA(A0.a1, r1, 0, hi);

  int cNext = 1, cStage = 2;
  for (int s = 0; s < STEPS; ++s) {
    if (s + 2 < STEPS) cpB(stBase + cStage * PANEL, bsrc + (size_t)(s + 2) * 32);
    WMMA8(A0);
    if (s + 1 < STEPS) {
      loadA(A0.a0, r0, (s + 1) * 32, hi);
      loadA(A0.a1, r1, (s + 1) * 32, hi);
    }
    if (s + 2 < STEPS) asyncWait1(); else asyncWait0();
    wgBarrier();
    if (s + 1 < STEPS) DSLOAD_B(&Bsh[0][0] + (size_t)cNext * PANEL);
    if (++cNext >= 3) cNext = 0;
    if (++cStage >= 3) cStage = 0;
  }

  float bv0 = bias[Nb + lo], bv1 = bias[Nb + 16 + lo];
  float bv2 = bias[Nb + 32 + lo], bv3 = bias[Nb + 48 + lo];
#pragma unroll
  for (int r = 0; r < 8; ++r) {
    int row0 = Mb + hi * 8 + r;
    unsigned short* o0 = out + (size_t)row0 * D2 + Nb + lo;
    unsigned short* o1 = out + (size_t)(row0 + 16) * D2 + Nb + lo;
    float t;
    t = acc0[r] + bv0; o0[0]  = f2bf(t > 0.f ? t : 0.f);
    t = acc1[r] + bv1; o0[16] = f2bf(t > 0.f ? t : 0.f);
    t = acc2[r] + bv2; o0[32] = f2bf(t > 0.f ? t : 0.f);
    t = acc3[r] + bv3; o0[48] = f2bf(t > 0.f ? t : 0.f);
    t = acc4[r] + bv0; o1[0]  = f2bf(t > 0.f ? t : 0.f);
    t = acc5[r] + bv1; o1[16] = f2bf(t > 0.f ? t : 0.f);
    t = acc6[r] + bv2; o1[32] = f2bf(t > 0.f ? t : 0.f);
    t = acc7[r] + bv3; o1[48] = f2bf(t > 0.f ? t : 0.f);
  }
}

// ---------------- conv cumsum: flat 288-step pipeline (12 taps x 24 K-steps);
// store relu(acc) every 24th step, accumulators carry the cumulative sum.
__global__ __launch_bounds__(256) void conv_cumsum(
    const unsigned short* __restrict__ H,   // BL x D
    const unsigned short* __restrict__ CW,  // W x D x D (k,o,c)
    unsigned short* __restrict__ out)       // ROWS x D relu'd bf16
{
  __shared__ unsigned short Bsh[3][PANEL];
  const int NB = D_ / 64;                                  // 12 N-blocks
  int nb = blockIdx.x % NB, mb = blockIdx.x / NB;
  int Nb = nb * 64;
  int w  = threadIdx.x >> 5;
  int lane = threadIdx.x & 31, lo = lane & 15, hi = lane >> 4;
  int Mb = mb * 256 + w * 32;
  int bb = Mb / L_, l0 = Mb % L_;                          // never crosses batch

  int srow = threadIdx.x >> 2, schunk = threadIdx.x & 3;
  const unsigned short* bsrcBase = CW + (size_t)(Nb + srow) * D_ + schunk * 8;
  unsigned short* stBase = &Bsh[0][0] + srow * LDB_LD + schunk * 8;

  v8f acc0={},acc1={},acc2={},acc3={},acc4={},acc5={},acc6={},acc7={};
  Frag bf0, bf1, bf2, bf3;
  ASet A0;
  const int STEPS = W_ * (D_ / 32);                        // 288

  // staging source for flat step s: k-slice s/24, K-chunk (s%24)*32
  cpB(stBase + 0 * PANEL, bsrcBase);                       // s=0: k=0, kk=0
  cpB(stBase + 1 * PANEL, bsrcBase + 32);                  // s=1: k=0, kk=32
  asyncWait1();
  wgBarrier();
  DSLOAD_B(&Bsh[0][0]);
  {                                                        // A for step 0 (k=0)
    int sl0 = l0 + lo, sl1 = l0 + 16 + lo;                 // always < L_ at k=0
    loadA(A0.a0, H + (size_t)(bb * L_ + sl0) * D_, 0, hi);
    loadA(A0.a1, H + (size_t)(bb * L_ + sl1) * D_, 0, hi);
  }

  int cNext = 1, cStage = 2;
  for (int s = 0; s < STEPS; ++s) {
    if (s + 2 < STEPS) {
      int ss = s + 2, kq = ss / 24;
      cpB(stBase + cStage * PANEL,
          bsrcBase + (size_t)kq * (D_ * D_) + (size_t)(ss - kq * 24) * 32);
    }
    WMMA8(A0);
    if ((s % 24) == 23) {                                  // tap kq complete
      int kq = s / 24;
#pragma unroll
      for (int r = 0; r < 8; ++r) {
        int lr0 = l0 + hi * 8 + r;
        unsigned short* o0 = out + ((size_t)(bb * L_ + lr0) * W_ + kq) * D_ + Nb + lo;
        unsigned short* o1 = out + ((size_t)(bb * L_ + lr0 + 16) * W_ + kq) * D_ + Nb + lo;
        float t;
        t = acc0[r]; o0[0]  = f2bf(t > 0.f ? t : 0.f);
        t = acc1[r]; o0[16] = f2bf(t > 0.f ? t : 0.f);
        t = acc2[r]; o0[32] = f2bf(t > 0.f ? t : 0.f);
        t = acc3[r]; o0[48] = f2bf(t > 0.f ? t : 0.f);
        t = acc4[r]; o1[0]  = f2bf(t > 0.f ? t : 0.f);
        t = acc5[r]; o1[16] = f2bf(t > 0.f ? t : 0.f);
        t = acc6[r]; o1[32] = f2bf(t > 0.f ? t : 0.f);
        t = acc7[r]; o1[48] = f2bf(t > 0.f ? t : 0.f);
      }
    }
    if (s + 1 < STEPS) {                                   // A for step s+1
      int ss = s + 1, kq = ss / 24, kk = (ss - kq * 24) * 32;
      int sl0 = l0 + lo + kq, sl1 = l0 + 16 + lo + kq;
      bool v0 = sl0 < L_, v1 = sl1 < L_;
      loadA(A0.a0, H + (size_t)(bb * L_ + (v0 ? sl0 : 0)) * D_, kk, hi);
      if (!v0) { A0.a0.q[0] = U4{0,0,0,0}; A0.a0.q[1] = U4{0,0,0,0}; }
      loadA(A0.a1, H + (size_t)(bb * L_ + (v1 ? sl1 : 0)) * D_, kk, hi);
      if (!v1) { A0.a1.q[0] = U4{0,0,0,0}; A0.a1.q[1] = U4{0,0,0,0}; }
    }
    if (s + 2 < STEPS) asyncWait1(); else asyncWait0();
    wgBarrier();
    if (s + 1 < STEPS) DSLOAD_B(&Bsh[0][0] + (size_t)cNext * PANEL);
    if (++cNext >= 3) cNext = 0;
    if (++cStage >= 3) cStage = 0;
  }
}

// ---- final: out = relu([g_start|g_end|conv] @ out_w^T + b), gather fused in A
__global__ __launch_bounds__(256) void out_gemm(
    const unsigned short* __restrict__ P,   // BL x D2 relu'd proj
    const unsigned short* __restrict__ C,   // ROWS x D relu'd conv
    const long long* __restrict__ SI,       // ROWS x 2 (int64)
    const unsigned short* __restrict__ OW,  // D x K3 (N x K)
    const float* __restrict__ bias,         // D
    float* __restrict__ out)                // ROWS x D
{
  __shared__ unsigned short Bsh[3][PANEL];
  const int NB = D_ / 64;                                  // 12 N-blocks
  int nb = blockIdx.x % NB, mb = blockIdx.x / NB;
  int Nb = nb * 64;
  int w  = threadIdx.x >> 5;
  int lane = threadIdx.x & 31, lo = lane & 15, hi = lane >> 4;
  int Mb = mb * 256 + w * 32;

  const unsigned short* segA[2][3];
#pragma unroll
  for (int t = 0; t < 2; ++t) {
    int m  = Mb + t * 16 + lo;
    int bb = m / (L_ * W_);
    long long s = SI[(size_t)m * 2 + 0];
    long long e = SI[(size_t)m * 2 + 1];
    s = s < 0 ? 0 : (s > L_ - 1 ? L_ - 1 : s);
    e = e < 0 ? 0 : (e > L_ - 1 ? L_ - 1 : e);
    segA[t][0] = P + (size_t)(bb * L_ + (int)s) * D2;      // start_rep [0,768)
    segA[t][1] = P + (size_t)(bb * L_ + (int)e) * D2 + D_; // end_rep   [768,1536)
    segA[t][2] = C + (size_t)m * D_;                       // conv_span
  }

  int srow = threadIdx.x >> 2, schunk = threadIdx.x & 3;
  const unsigned short* bsrc = OW + (size_t)(Nb + srow) * K3 + schunk * 8;
  unsigned short* stBase = &Bsh[0][0] + srow * LDB_LD + schunk * 8;

  v8f acc0={},acc1={},acc2={},acc3={},acc4={},acc5={},acc6={},acc7={};
  Frag bf0, bf1, bf2, bf3;
  ASet A0;
  const int STEPS = K3 / 32;                               // 72

  cpB(stBase + 0 * PANEL, bsrc);
  cpB(stBase + 1 * PANEL, bsrc + 32);
  asyncWait1();
  wgBarrier();
  DSLOAD_B(&Bsh[0][0]);
  loadA(A0.a0, segA[0][0], 0, hi);
  loadA(A0.a1, segA[1][0], 0, hi);

  int cNext = 1, cStage = 2;
  for (int s = 0; s < STEPS; ++s) {
    if (s + 2 < STEPS) cpB(stBase + cStage * PANEL, bsrc + (size_t)(s + 2) * 32);
    WMMA8(A0);
    if (s + 1 < STEPS) {
      int ss = s + 1, seg = ss / 24, kk = (ss - seg * 24) * 32;
      loadA(A0.a0, segA[0][seg], kk, hi);
      loadA(A0.a1, segA[1][seg], kk, hi);
    }
    if (s + 2 < STEPS) asyncWait1(); else asyncWait0();
    wgBarrier();
    if (s + 1 < STEPS) DSLOAD_B(&Bsh[0][0] + (size_t)cNext * PANEL);
    if (++cNext >= 3) cNext = 0;
    if (++cStage >= 3) cStage = 0;
  }

  float bv0 = bias[Nb + lo], bv1 = bias[Nb + 16 + lo];
  float bv2 = bias[Nb + 32 + lo], bv3 = bias[Nb + 48 + lo];
#pragma unroll
  for (int r = 0; r < 8; ++r) {
    int row0 = Mb + hi * 8 + r;
    float* o0 = out + (size_t)row0 * D_ + Nb + lo;
    float* o1 = out + (size_t)(row0 + 16) * D_ + Nb + lo;
    float t;
    t = acc0[r] + bv0; o0[0]  = t > 0.f ? t : 0.f;
    t = acc1[r] + bv1; o0[16] = t > 0.f ? t : 0.f;
    t = acc2[r] + bv2; o0[32] = t > 0.f ? t : 0.f;
    t = acc3[r] + bv3; o0[48] = t > 0.f ? t : 0.f;
    t = acc4[r] + bv0; o1[0]  = t > 0.f ? t : 0.f;
    t = acc5[r] + bv1; o1[16] = t > 0.f ? t : 0.f;
    t = acc6[r] + bv2; o1[32] = t > 0.f ? t : 0.f;
    t = acc7[r] + bv3; o1[48] = t > 0.f ? t : 0.f;
  }
}

extern "C" void kernel_launch(void* const* d_in, const int* in_sizes, int n_in,
                              void* d_out, int out_size, void* d_ws, size_t ws_size,
                              hipStream_t stream) {
  const float*     h       = (const float*)d_in[0];
  const long long* span_ix = (const long long*)d_in[1];   // int64 in reference
  const float*     proj_w  = (const float*)d_in[2];
  const float*     proj_b  = (const float*)d_in[3];
  const float*     conv_w  = (const float*)d_in[4];
  const float*     out_w   = (const float*)d_in[5];
  const float*     out_b   = (const float*)d_in[6];
  float*           out     = (float*)d_out;

  unsigned short* ws = (unsigned short*)d_ws;
  size_t off = 0;
  auto carve = [&](size_t n) { size_t o = off; off += (n + 127) & ~(size_t)127; return o; };
  unsigned short* h_bf  = ws + carve((size_t)BL * D_);      //  3.0 MB
  unsigned short* pw_bf = ws + carve((size_t)D2 * D_);      //  2.3 MB
  unsigned short* ow_bf = ws + carve((size_t)D_ * K3);      //  3.4 MB
  unsigned short* cw_bf = ws + carve((size_t)W_ * D_ * D_); // 13.5 MB
  unsigned short* projR = ws + carve((size_t)BL * D2);      //  6.3 MB
  unsigned short* convR = ws + carve((size_t)ROWS * D_);    // 36.0 MB

  cvt_f32_bf16<<<512, 256, 0, stream>>>(h,      h_bf,  BL * D_);
  cvt_f32_bf16<<<512, 256, 0, stream>>>(proj_w, pw_bf, D2 * D_);
  cvt_f32_bf16<<<512, 256, 0, stream>>>(out_w,  ow_bf, D_ * K3);
  cvt_convw  <<<1024, 256, 0, stream>>>(conv_w, cw_bf);

  gemm_proj  <<<(BL / 256) * (D2 / 64), 256, 0, stream>>>(h_bf, pw_bf, proj_b, projR);
  conv_cumsum<<<(BL / 256) * (D_ / 64), 256, 0, stream>>>(h_bf, cw_bf, convR);
  out_gemm   <<<(ROWS / 256) * (D_ / 64), 256, 0, stream>>>(projR, convR, span_ix,
                                                            ow_bf, out_b, out);
}
